// CausalUpsamplingLRU_54898271977944
// MI455X (gfx1250) — matrix operand order repaired and checked by reading, not compiled, not run on hardware
//
#include <hip/hip_runtime.h>

// ---------------------------------------------------------------------------
// CausalUpsamplingLRU for gfx1250 (MI455X), wave32 + v_wmma_f32_16x16x32_bf16
//
// Shapes: IN_CH=OUT_CH=256, STATE=384, B=8, SEQ=512, OUT_SEQ=32.  M = B*SEQ = 4096.
// Per pass:  U[4096,768]  = Xcur[4096,256] @ Bg[768,256]^T          (gamma folded in Bg)
//            scan over S  : h = lam*h + u   (f32 state, bf16 hs out)
//            Y [4096,256] = [HS|Xcur][4096,1024] @ Wpk[256,1024]^T  (C_re|-C_im|D packed)
//
// GEMMs use 32x32 per-wave register blocking: 2 A-frags x 2 B-frags -> 4 wmma
// per k-step (16 FLOP/byte of fragment traffic, 2x the naive 16x16 tiling).
// ---------------------------------------------------------------------------

typedef __attribute__((ext_vector_type(16))) __bf16 v16bf;
typedef __attribute__((ext_vector_type(8)))  float  v8f;

#define M_ROWS   4096      // B_SZ * SEQ
#define N_STATE  384
#define N_U      768       // re|im stacked
#define K_IN     256
#define K_PACK   1024      // 384 + 384 + 256
#define SEQ_LEN  512
#define B_SZ     8
#define OUT_SEQ  32

static __device__ __forceinline__ __bf16 f2b(float f) {
    unsigned u = __float_as_uint(f);
    unsigned r = (u + 0x7FFFu + ((u >> 16) & 1u)) >> 16;   // round-to-nearest-even
    unsigned short s = (unsigned short)r;
    __bf16 b;
    __builtin_memcpy(&b, &s, 2);
    return b;
}

// A-fragment (16x32 bf16): lane holds row m0+(lane&15); two 16B chunks at
// K = k0 + 8*(lane>>4) and K = k0 + 16 + 8*(lane>>4).
static __device__ __forceinline__ v16bf load_frag_a(const __bf16* __restrict__ rowptr,
                                                    int k0, int lane) {
    const __bf16* p = rowptr + k0 + ((lane >> 4) << 3);
    union { v16bf v; uint4 q[2]; } u;
    u.q[0] = *reinterpret_cast<const uint4*>(p);
    u.q[1] = *reinterpret_cast<const uint4*>(p + 16);
    return u.v;
}

// B-fragment (32x16 bf16) from row-major W[N,K]: lane holds column n0+(lane&15);
// two 16B chunks at K = k0 + 8*(lane>>4) and K = k0 + 16 + 8*(lane>>4).
static __device__ __forceinline__ v16bf load_frag_b(const __bf16* __restrict__ rowptr,
                                                    int k0, int lane) {
    const __bf16* p = rowptr + k0 + ((lane >> 4) << 3);
    union { v16bf v; uint4 q[2]; } u;
    u.q[0] = *reinterpret_cast<const uint4*>(p);
    u.q[1] = *reinterpret_cast<const uint4*>(p + 16);
    return u.v;
}

#define WMMA_BF16(A, B, C) \
    __builtin_amdgcn_wmma_f32_16x16x32_bf16(false, (A), false, (B), (short)0, (C), false, false)

// ---------------------------------------------------------------------------
// Prep: fold gamma into B rows -> Bg[768,256] bf16; pack [C_re | -C_im | D]
// -> Wpk[256,1024] bf16; compute lambda re/im; zero H.  262144 threads.
// ---------------------------------------------------------------------------
__global__ __launch_bounds__(256) void lru_prep_w(
    const float* __restrict__ nu_log, const float* __restrict__ th_log,
    const float* __restrict__ ga_log,
    const float* __restrict__ B_re, const float* __restrict__ B_im,
    const float* __restrict__ C_re, const float* __restrict__ C_im,
    const float* __restrict__ Dm,
    __bf16* __restrict__ Bg, __bf16* __restrict__ Wpk,
    float* __restrict__ lre, float* __restrict__ lim,
    float* __restrict__ Hre, float* __restrict__ Him)
{
    int i = blockIdx.x * 256 + threadIdx.x;            // 0 .. 262143

    if (i < N_STATE) {
        float lm = __expf(-__expf(nu_log[i]));
        float th = __expf(th_log[i]);
        lre[i] = lm * __cosf(th);
        lim[i] = lm * __sinf(th);
    }
    if (i < B_SZ * N_STATE) { Hre[i] = 0.f; Him[i] = 0.f; }

    if (i < N_U * K_IN) {                              // Bg: gamma folded
        int n = i >> 8, c = i & 255;
        int ns = (n < N_STATE) ? n : n - N_STATE;
        float g = __expf(ga_log[ns]);
        float v = g * ((n < N_STATE) ? B_re[ns * K_IN + c] : B_im[ns * K_IN + c]);
        Bg[i] = f2b(v);
    }
    {                                                  // Wpk: exactly 262144 elems
        int n = i >> 10, k = i & 1023;
        float v;
        if      (k < N_STATE)     v =  C_re[n * N_STATE + k];
        else if (k < 2 * N_STATE) v = -C_im[n * N_STATE + (k - N_STATE)];
        else                      v =  Dm[n * K_IN + (k - 2 * N_STATE)];
        Wpk[i] = f2b(v);
    }
}

__global__ __launch_bounds__(256) void lru_prep_x(const float* __restrict__ x,
                                                  __bf16* __restrict__ X0)
{
    int i = blockIdx.x * 256 + threadIdx.x;            // 0 .. 1048575
    X0[i] = f2b(x[i]);
}

// ---------------------------------------------------------------------------
// GEMM 1:  U[4096,768] (f32) = Xcur[4096,256] @ Bg[768,256]^T
// Wave computes 32x32; block = 8 waves (2x4) -> 64x128 tile.  grid = (64, 6)
// ---------------------------------------------------------------------------
__global__ __launch_bounds__(256) void lru_gemm_u(
    const __bf16* __restrict__ X, const __bf16* __restrict__ Bg,
    float* __restrict__ U)
{
    const int lane = threadIdx.x & 31;
    const int wave = threadIdx.x >> 5;
    const int m0 = (blockIdx.x * 2 + (wave >> 2)) * 32;
    const int n0 = (blockIdx.y * 4 + (wave & 3)) * 32;

    const __bf16* arow0 = X  + (size_t)(m0 +      (lane & 15)) * K_IN;
    const __bf16* arow1 = X  + (size_t)(m0 + 16 + (lane & 15)) * K_IN;
    const __bf16* brow0 = Bg + (size_t)(n0 +      (lane & 15)) * K_IN;
    const __bf16* brow1 = Bg + (size_t)(n0 + 16 + (lane & 15)) * K_IN;

    v8f acc00 = {}, acc01 = {}, acc10 = {}, acc11 = {};
#pragma unroll
    for (int k0 = 0; k0 < K_IN; k0 += 32) {
        v16bf a0 = load_frag_a(arow0, k0, lane);
        v16bf a1 = load_frag_a(arow1, k0, lane);
        v16bf b0 = load_frag_b(brow0, k0, lane);
        v16bf b1 = load_frag_b(brow1, k0, lane);
        acc00 = WMMA_BF16(a0, b0, acc00);
        acc01 = WMMA_BF16(a0, b1, acc01);
        acc10 = WMMA_BF16(a1, b0, acc10);
        acc11 = WMMA_BF16(a1, b1, acc11);
    }
    const int orow = m0 + ((lane >> 4) << 3);
    const int ocol = n0 + (lane & 15);
#pragma unroll
    for (int r = 0; r < 8; ++r) {
        U[(size_t)(orow + r)      * N_U + ocol]      = acc00[r];
        U[(size_t)(orow + r)      * N_U + ocol + 16] = acc01[r];
        U[(size_t)(orow + r + 16) * N_U + ocol]      = acc10[r];
        U[(size_t)(orow + r + 16) * N_U + ocol + 16] = acc11[r];
    }
}

// ---------------------------------------------------------------------------
// Scan: 3072 threads, one per (b, n).  f32 recurrence state, bf16 hs output
// packed as HS[m, 0:384]=re, HS[m, 384:768]=im.
// ---------------------------------------------------------------------------
__global__ __launch_bounds__(256) void lru_scan(
    const float* __restrict__ U,
    const float* __restrict__ lre, const float* __restrict__ lim,
    float* __restrict__ Hre, float* __restrict__ Him,
    __bf16* __restrict__ HS)
{
    int t = blockIdx.x * 256 + threadIdx.x;            // 0 .. 3071
    int b = t / N_STATE, n = t - b * N_STATE;

    float lr = lre[n], li = lim[n];
    float hr = Hre[t], hi = Him[t];

    const float* Ub = U  + (size_t)b * SEQ_LEN * N_U;
    __bf16*      Hb = HS + (size_t)b * SEQ_LEN * N_U;

    for (int s = 0; s < SEQ_LEN; ++s) {
        float ur = Ub[(size_t)s * N_U + n];
        float ui = Ub[(size_t)s * N_U + N_STATE + n];
        float nr = fmaf(lr, hr, fmaf(-li, hi, ur));
        float ni = fmaf(lr, hi, fmaf( li, hr, ui));
        hr = nr; hi = ni;
        Hb[(size_t)s * N_U + n]           = f2b(hr);
        Hb[(size_t)s * N_U + N_STATE + n] = f2b(hi);
    }
    Hre[t] = hr; Him[t] = hi;
}

// ---------------------------------------------------------------------------
// GEMM 2:  Y[4096,256] = HS[4096,768] @ Wpk[:,0:768]^T + Xcur[4096,256] @ Wpk[:,768:]^T
// Wave computes 32x32; block = 8 waves (2x4) -> 64x128 tile.  grid = (64, 2)
// Writes bf16 Y -> Xnext (next pass input) and the s==511 rows (f32) -> out[b, t, :].
// ---------------------------------------------------------------------------
__global__ __launch_bounds__(256) void lru_gemm_y(
    const __bf16* __restrict__ HS, const __bf16* __restrict__ Xc,
    const __bf16* __restrict__ Wpk,
    __bf16* __restrict__ Xn, float* __restrict__ out, int iter)
{
    const int lane = threadIdx.x & 31;
    const int wave = threadIdx.x >> 5;
    const int m0 = (blockIdx.x * 2 + (wave >> 2)) * 32;
    const int n0 = (blockIdx.y * 4 + (wave & 3)) * 32;

    const __bf16* hrow0 = HS  + (size_t)(m0 +      (lane & 15)) * N_U;
    const __bf16* hrow1 = HS  + (size_t)(m0 + 16 + (lane & 15)) * N_U;
    const __bf16* xrow0 = Xc  + (size_t)(m0 +      (lane & 15)) * K_IN;
    const __bf16* xrow1 = Xc  + (size_t)(m0 + 16 + (lane & 15)) * K_IN;
    const __bf16* wrow0 = Wpk + (size_t)(n0 +      (lane & 15)) * K_PACK;
    const __bf16* wrow1 = Wpk + (size_t)(n0 + 16 + (lane & 15)) * K_PACK;

    v8f acc00 = {}, acc01 = {}, acc10 = {}, acc11 = {};
#pragma unroll
    for (int k0 = 0; k0 < N_U; k0 += 32) {             // hs_re | hs_im segment
        v16bf a0 = load_frag_a(hrow0, k0, lane);
        v16bf a1 = load_frag_a(hrow1, k0, lane);
        v16bf b0 = load_frag_b(wrow0, k0, lane);
        v16bf b1 = load_frag_b(wrow1, k0, lane);
        acc00 = WMMA_BF16(a0, b0, acc00);
        acc01 = WMMA_BF16(a0, b1, acc01);
        acc10 = WMMA_BF16(a1, b0, acc10);
        acc11 = WMMA_BF16(a1, b1, acc11);
    }
#pragma unroll
    for (int k0 = 0; k0 < K_IN; k0 += 32) {            // x segment (D term)
        v16bf a0 = load_frag_a(xrow0, k0, lane);
        v16bf a1 = load_frag_a(xrow1, k0, lane);
        v16bf b0 = load_frag_b(wrow0 + N_U, k0, lane);
        v16bf b1 = load_frag_b(wrow1 + N_U, k0, lane);
        acc00 = WMMA_BF16(a0, b0, acc00);
        acc01 = WMMA_BF16(a0, b1, acc01);
        acc10 = WMMA_BF16(a1, b0, acc10);
        acc11 = WMMA_BF16(a1, b1, acc11);
    }

    const int orow = m0 + ((lane >> 4) << 3);
    const int ocol = n0 + (lane & 15);
#pragma unroll
    for (int r = 0; r < 8; ++r) {
        int m_lo = orow + r;
        int m_hi = orow + r + 16;
        float v00 = acc00[r], v01 = acc01[r];
        float v10 = acc10[r], v11 = acc11[r];
        Xn[(size_t)m_lo * K_IN + ocol]      = f2b(v00);
        Xn[(size_t)m_lo * K_IN + ocol + 16] = f2b(v01);
        Xn[(size_t)m_hi * K_IN + ocol]      = f2b(v10);
        Xn[(size_t)m_hi * K_IN + ocol + 16] = f2b(v11);
        if ((m_lo & (SEQ_LEN - 1)) == SEQ_LEN - 1) {   // last timestep of a batch row
            int b = m_lo >> 9;
            out[(size_t)(b * OUT_SEQ + iter) * K_IN + ocol]      = v00;
            out[(size_t)(b * OUT_SEQ + iter) * K_IN + ocol + 16] = v01;
        }
        if ((m_hi & (SEQ_LEN - 1)) == SEQ_LEN - 1) {
            int b = m_hi >> 9;
            out[(size_t)(b * OUT_SEQ + iter) * K_IN + ocol]      = v10;
            out[(size_t)(b * OUT_SEQ + iter) * K_IN + ocol + 16] = v11;
        }
    }
}

// ---------------------------------------------------------------------------
extern "C" void kernel_launch(void* const* d_in, const int* in_sizes, int n_in,
                              void* d_out, int out_size, void* d_ws, size_t ws_size,
                              hipStream_t stream) {
    const float* x      = (const float*)d_in[0];
    const float* nu_log = (const float*)d_in[1];
    const float* th_log = (const float*)d_in[2];
    const float* ga_log = (const float*)d_in[3];
    const float* B_re   = (const float*)d_in[4];
    const float* B_im   = (const float*)d_in[5];
    const float* C_re   = (const float*)d_in[6];
    const float* C_im   = (const float*)d_in[7];
    const float* Dm     = (const float*)d_in[8];
    float* out = (float*)d_out;

    char* w = (char*)d_ws;
    size_t o = 0;
    auto take = [&](size_t bytes) { char* p = w + o; o = (o + bytes + 255) & ~(size_t)255; return p; };

    __bf16* HS  = (__bf16*)take((size_t)M_ROWS * N_U * 2);      //  6.0 MB
    __bf16* X0  = (__bf16*)take((size_t)M_ROWS * K_IN * 2);     //  2.0 MB
    __bf16* X1  = (__bf16*)take((size_t)M_ROWS * K_IN * 2);     //  2.0 MB
    float*  U   = (float*) take((size_t)M_ROWS * N_U * 4);      // 12.0 MB
    __bf16* Bg  = (__bf16*)take((size_t)N_U * K_IN * 2);
    __bf16* Wpk = (__bf16*)take((size_t)K_IN * K_PACK * 2);
    float*  lre = (float*) take(N_STATE * 4);
    float*  lim = (float*) take(N_STATE * 4);
    float*  Hre = (float*) take(B_SZ * N_STATE * 4);
    float*  Him = (float*) take(B_SZ * N_STATE * 4);

    // one-time prep (runs every launch -> deterministic)
    lru_prep_w<<<1024, 256, 0, stream>>>(nu_log, th_log, ga_log, B_re, B_im,
                                         C_re, C_im, Dm, Bg, Wpk, lre, lim, Hre, Him);
    lru_prep_x<<<(M_ROWS * K_IN) / 256, 256, 0, stream>>>(x, X0);

    for (int t = 0; t < OUT_SEQ; ++t) {
        __bf16* Xc = (t & 1) ? X1 : X0;
        __bf16* Xn = (t & 1) ? X0 : X1;
        lru_gemm_u<<<dim3(M_ROWS / 64, N_U / 128), 256, 0, stream>>>(Xc, Bg, U);
        lru_scan  <<<(B_SZ * N_STATE) / 256, 256, 0, stream>>>(U, lre, lim, Hre, Him, HS);
        lru_gemm_y<<<dim3(M_ROWS / 64, K_IN / 128), 256, 0, stream>>>(HS, Xc, Wpk, Xn, out, t);
    }
}
